// MoE_15917148799372
// MI455X (gfx1250) — compile-verified, matrix operand order checked by reading, not run on hardware
//
#include <hip/hip_runtime.h>
#include <cstdint>
#include <cstddef>

#define D_MODELc 1024
#define D_FFc    4096
#define NEXP     8
#define TTOK     8192   // B*L = 4*2048
#define HLD      40     // H LDS row stride (bf16 elems), mult of 8
#define XLD      1032   // X / W1t LDS row stride (bf16 elems), mult of 8
#define W2LD     40     // W2t LDS row stride (bf16 elems), mult of 8
#define NCHUNK   (D_FFc / 32)   // 128 d_ff chunks

typedef __attribute__((ext_vector_type(16))) __bf16 v16bf;
typedef __attribute__((ext_vector_type(8)))  float  v8f;

#if __has_builtin(__builtin_amdgcn_tensor_load_to_lds) && \
    __has_builtin(__builtin_amdgcn_s_wait_tensorcnt)
#define HAVE_TDM 1
typedef unsigned tdm_u32x4 __attribute__((ext_vector_type(4)));
typedef int      tdm_i32x8 __attribute__((ext_vector_type(8)));
typedef int      tdm_i32x4 __attribute__((ext_vector_type(4)));
#else
#define HAVE_TDM 0
#endif

// ---------------- helpers ----------------

// Pack two f32 -> two bf16 (round-half-up) in one dword via V_PERM_B32.
__device__ __forceinline__ unsigned pack2_bf16(float lo, float hi) {
    unsigned a = __builtin_bit_cast(unsigned, lo) + 0x8000u;
    unsigned b = __builtin_bit_cast(unsigned, hi) + 0x8000u;
    return __builtin_amdgcn_perm(b, a, 0x07060302u);  // {hi16(b), hi16(a)}
}

__device__ __forceinline__ unsigned short f32_to_bf16_bits(float f) {
    unsigned u = __builtin_bit_cast(unsigned, f) + 0x8000u;
    return (unsigned short)(u >> 16);
}

// Load 8 k-strided f32, convert to bf16, store one b128 (LDS or global dst).
__device__ __forceinline__ void stage8_kstride(const float* src, size_t stride,
                                               unsigned short* dst) {
    float f[8];
#pragma unroll
    for (int i = 0; i < 8; ++i) f[i] = src[(size_t)i * stride];
    uint4 pk;
    pk.x = pack2_bf16(f[0], f[1]);
    pk.y = pack2_bf16(f[2], f[3]);
    pk.z = pack2_bf16(f[4], f[5]);
    pk.w = pack2_bf16(f[6], f[7]);
    *(uint4*)dst = pk;
}

// A/B fragment for V_WMMA_F32_16X16X32_BF16 from an LDS row-major matrix.
// 16-bit A layout (ISA 7.12.2): lane<16: halves 0..7 -> K=k0..k0+7, halves
// 8..15 -> K=k0+16..k0+23 ; lane>=16: +8 on both runs. Two contiguous 16B runs.
__device__ __forceinline__ v16bf lds_frag(const unsigned short* p, int row,
                                          int ld, int k0, int laneHi8) {
    union { v16bf v; uint4 q[2]; } u;
    const unsigned short* r = p + row * ld + k0 + laneHi8;
    u.q[0] = *(const uint4*)(r);
    u.q[1] = *(const uint4*)(r + 16);
    return u.v;
}

__device__ __forceinline__ v8f wmma_bf16(v16bf a, v16bf b, v8f c) {
    return __builtin_amdgcn_wmma_f32_16x16x32_bf16(false, a, false, b,
                                                   (short)0, c, false, false);
}

#if HAVE_TDM
// TDM: DMA one 64 KB bf16 W2 chunk (32768 elems) from global into LDS,
// inserting 16 B of padding after every 64 B row -> row stride W2LD=40 elems.
// D# bitfields per CDNA5 ISA (1-D tile, <=2D).
__device__ __forceinline__ void tdm_load_w2_chunk(const unsigned short* gsrc,
                                                  unsigned lds_off) {
    unsigned long long ga = (unsigned long long)(uintptr_t)gsrc;
    tdm_u32x4 g0;
    g0.x = 1u;                                             // count=1, user desc
    g0.y = lds_off;                                        // lds_addr (bytes)
    g0.z = (unsigned)(ga & 0xffffffffu);                   // global_addr[31:0]
    g0.w = (unsigned)((ga >> 32) & 0x01ffffffu) | (2u << 30); // addr[56:32]|type=2
    tdm_i32x8 g1;
    g1[0] = (1 << 16)      /* data_size: 2 bytes      */
          | (1 << 20)      /* pad_enable              */
          | (3 << 22)      /* pad_interval: 16 DWORDs */
          | (3 << 25);     /* pad_amount:   4 DWORDs  */
    g1[1] = (int)0x80000000u;   // tensor_dim0 = 32768 (low16 -> bits[63:48])
    g1[2] = 0;                  // tensor_dim0 hi16 / tensor_dim1 lo16
    g1[3] = (int)0x80000000u;   // tile_dim0 = 32768 -> bits[127:112]
    g1[4] = 0;                  // tile_dim1 = 0 (1-D), tile_dim2 = 0
    g1[5] = 32768;              // tensor_dim0_stride (unused for 1-D)
    g1[6] = 0;
    g1[7] = 0;
    tdm_i32x4 z4 = {0, 0, 0, 0};
    tdm_i32x8 z8 = {0, 0, 0, 0, 0, 0, 0, 0};
    // 6-arg toolchain form: (g0, g1, g2, g3, extra, cpol)
    __builtin_amdgcn_tensor_load_to_lds(g0, g1, z4, z4, z8, 0);
}
#endif

// ---------------- small kernels ----------------

__global__ void moe_init_kernel(float* __restrict__ out, int* __restrict__ cnt,
                                float* __restrict__ probsum) {
    size_t i = (size_t)blockIdx.x * blockDim.x + threadIdx.x;  // one float4 each
    float4 z = {0.f, 0.f, 0.f, 0.f};
    ((float4*)out)[i] = z;
    if (blockIdx.x == 0 && threadIdx.x < NEXP) {
        cnt[threadIdx.x] = 0;
        probsum[threadIdx.x] = 0.f;
    }
}

// W1 [e][k][n] f32 -> w1bf [e][n][k] bf16 (k contiguous).
__global__ void moe_conv_w1_kernel(const float* __restrict__ W1,
                                   unsigned short* __restrict__ w1bf) {
    int row = blockIdx.x * 256 + threadIdx.x;   // 0 .. E*D_FF-1
    int e = row >> 12;
    int n = row & (D_FFc - 1);
    const float* src = W1 + (size_t)e * D_MODELc * D_FFc + n;
    unsigned short* dst = w1bf + (size_t)row * D_MODELc;
#pragma unroll 2
    for (int k = 0; k < D_MODELc; k += 8)
        stage8_kstride(src + (size_t)k * D_FFc, D_FFc, dst + k);
}

// W2 [e][k][n] f32 -> w2bf chunk-major [e][fc][n][32] bf16.
__global__ void moe_conv_w2_kernel(const float* __restrict__ W2,
                                   unsigned short* __restrict__ w2bf) {
    int idx = blockIdx.x * 256 + threadIdx.x;   // 0 .. E*4*1024-1
    int n    = idx & (D_MODELc - 1);
    int rest = idx >> 10;
    int e    = rest >> 2;
    int kc   = (rest & 3) * 1024;               // k-range of 1024
    const float* src = W2 + (size_t)e * D_FFc * D_MODELc + n;
#pragma unroll 2
    for (int k = kc; k < kc + 1024; k += 8) {
        unsigned short* dst = w2bf +
            ((size_t)(e * NCHUNK + (k >> 5)) * D_MODELc + n) * 32 + (k & 31);
        stage8_kstride(src + (size_t)k * D_MODELc, D_MODELc, dst);
    }
}

__global__ void moe_router_kernel(const float* __restrict__ x,
                                  const float* __restrict__ Wr,
                                  const float* __restrict__ br,
                                  float* __restrict__ cw,
                                  int* __restrict__ cnt,
                                  float* __restrict__ probsum,
                                  int* __restrict__ lists) {
    __shared__ float sWr[D_MODELc * NEXP];   // 32 KB
    __shared__ float sps[NEXP];
    int tid = threadIdx.x;
    for (int i = tid; i < D_MODELc * NEXP; i += 256) sWr[i] = Wr[i];
    if (tid < NEXP) sps[tid] = 0.f;
    __syncthreads();

    int t = blockIdx.x * 256 + tid;
    float acc[NEXP];
#pragma unroll
    for (int e = 0; e < NEXP; ++e) acc[e] = br[e];
    const float* xr = x + (size_t)t * D_MODELc;
    for (int d = 0; d < D_MODELc; d += 4) {
        float4 xv = *(const float4*)(xr + d);
#pragma unroll
        for (int e = 0; e < NEXP; ++e) {
            acc[e] += xv.x * sWr[(d + 0) * NEXP + e] + xv.y * sWr[(d + 1) * NEXP + e]
                    + xv.z * sWr[(d + 2) * NEXP + e] + xv.w * sWr[(d + 3) * NEXP + e];
        }
    }
    float mx = acc[0];
#pragma unroll
    for (int e = 1; e < NEXP; ++e) mx = fmaxf(mx, acc[e]);
    float p[NEXP]; float s = 0.f;
#pragma unroll
    for (int e = 0; e < NEXP; ++e) { p[e] = __expf((acc[e] - mx) * (1.0f / 0.7f)); s += p[e]; }
    float inv = 1.0f / s;
#pragma unroll
    for (int e = 0; e < NEXP; ++e) { p[e] *= inv; atomicAdd(&sps[e], p[e]); }

    int i0 = 0; float v0 = p[0];
#pragma unroll
    for (int e = 1; e < NEXP; ++e) if (p[e] > v0) { v0 = p[e]; i0 = e; }
    int i1 = (i0 == 0) ? 1 : 0; float v1 = p[i1];
#pragma unroll
    for (int e = 0; e < NEXP; ++e) if (e != i0 && p[e] > v1) { v1 = p[e]; i1 = e; }
    float sn = v0 + v1 + 1e-6f;
    float w0 = v0 / sn, w1 = v1 / sn;
#pragma unroll
    for (int e = 0; e < NEXP; ++e)
        cw[(size_t)t * NEXP + e] = (e == i0) ? w0 : ((e == i1) ? w1 : 0.f);

    int p0 = atomicAdd(&cnt[i0], 1); lists[i0 * TTOK + p0] = t;
    int p1 = atomicAdd(&cnt[i1], 1); lists[i1 * TTOK + p1] = t;

    __syncthreads();
    if (tid < NEXP) atomicAdd(&probsum[tid], sps[tid]);
}

__global__ void moe_aux_kernel(const int* __restrict__ cnt,
                               const float* __restrict__ probsum,
                               float* __restrict__ out) {
    if (threadIdx.x == 0) {
        float s = 0.f;
        for (int e = 0; e < NEXP; ++e)
            s += ((float)cnt[e] / (float)TTOK) * (probsum[e] / (float)TTOK);
        out[(size_t)TTOK * D_MODELc] = (float)NEXP * s;
    }
}

// Fused expert FFN: out[t] += cw * (silu(x W1 + b1) W2 + b2), chunked over d_ff.
// Block = 32 tokens of one expert, 256 threads = 8 wave32.
// PRECONV=true : weights already bf16; W2 chunk DMA'd by the TDM (if available).
// PRECONV=false: convert f32 weights on the fly while staging.
template <bool PRECONV>
__global__ void __launch_bounds__(256)
moe_ffn_kernel(const float* __restrict__ x,
               const float* __restrict__ W1, const float* __restrict__ b1,
               const float* __restrict__ W2, const float* __restrict__ b2,
               const float* __restrict__ cw, const int* __restrict__ cnt,
               const int* __restrict__ lists, float* __restrict__ out,
               const unsigned short* __restrict__ w1bf,
               const unsigned short* __restrict__ w2bf) {
    int e    = blockIdx.x >> 8;      // 256 tiles per expert
    int tile = blockIdx.x & 255;
    int tid  = threadIdx.x;
    int lane = tid & 31;
    int w    = tid >> 5;             // wave id 0..7
    int count = cnt[e];
    if (tile * 32 >= count) return;  // uniform early-exit, before any barrier

    __shared__ unsigned short Xl[32 * XLD];         // x tile, bf16        (~66 KB)
    __shared__ unsigned short W1t[32 * XLD];        // W1 chunk^T [n][k]   (~66 KB)
    __shared__ unsigned short W2t[D_MODELc * W2LD]; // W2 chunk^T [n][k]   (80 KB)
    __shared__ unsigned short Hl[32 * HLD];         // silu() tile, bf16   (2.5 KB)
    __shared__ float pb[8 * 32 * 8];                // phase-1 partials    (8 KB)
    __shared__ float swt[32];
    __shared__ int   stk[32];

    if (tid < 32) {
        int idx = tile * 32 + tid;
        if (idx < count) {
            int t = lists[e * TTOK + idx];
            stk[tid] = t;
            swt[tid] = cw[(size_t)t * NEXP + e];
        } else { stk[tid] = 0; swt[tid] = 0.f; }
    }
    __syncthreads();

    // Stage 32 x 1024 token tile f32 -> bf16 LDS.
    {
        int r  = tid >> 3;
        int c0 = (tid & 7) * 128;
        const float* xr = x + (size_t)stk[r] * D_MODELc + c0;
        unsigned short* dst = Xl + r * XLD + c0;
#pragma unroll
        for (int i = 0; i < 128; i += 4) {
            float4 f = *(const float4*)(xr + i);
            uint2 pk;
            pk.x = pack2_bf16(f.x, f.y);
            pk.y = pack2_bf16(f.z, f.w);
            *(uint2*)(dst + i) = pk;
        }
    }
    __syncthreads();

    const float* w1e = W1 + (size_t)e * D_MODELc * D_FFc;
    const float* w2e = W2 + (size_t)e * D_FFc * D_MODELc;
    const unsigned short* w1be = w1bf + (size_t)e * D_FFc * D_MODELc;
    int laneHi8 = (lane & 16) ? 8 : 0;
    int l15     = lane & 15;

    const v8f vzero = {0.f, 0.f, 0.f, 0.f, 0.f, 0.f, 0.f, 0.f};
    v8f acc[2][8];
#pragma unroll
    for (int m = 0; m < 2; ++m)
#pragma unroll
        for (int nt = 0; nt < 8; ++nt) acc[m][nt] = vzero;

    const int m_t = (w & 3) >> 1;
    const int n_t = w & 1;
    const int kh  = w >> 2;
    const int arow = m_t * 16 + l15;
    const int brow = n_t * 16 + l15;

    // fallback-path W1 staging role: column n = tid&31, k-range of 128.
    const int sn  = tid & 31;
    const int skb = (tid >> 5) * 128;
    const float* w1col = w1e + sn;
    unsigned short* w1dst = W1t + sn * XLD;

    for (int fc = 0; fc < NCHUNK; ++fc) {
        int ff0 = fc * 32;

        if constexpr (PRECONV) {
#if HAVE_TDM
            // ---- W2 chunk via Tensor Data Mover (issue early, wave 0) ----
            if (w == 0)
                tdm_load_w2_chunk(w2bf + (size_t)(e * NCHUNK + fc) * D_MODELc * 32,
                                  (unsigned)(uintptr_t)(void*)W2t);
#endif
            // ---- W1 chunk: rows ff0..ff0+31 of w1bf [n][k], pure b128 copies ----
            {
                int nl = tid >> 3;
                int c0 = (tid & 7) * 128;
                const unsigned short* src = w1be + (size_t)(ff0 + nl) * D_MODELc + c0;
                unsigned short* dst = W1t + nl * XLD + c0;
#pragma unroll
                for (int i = 0; i < 128; i += 8)
                    *(uint4*)(dst + i) = *(const uint4*)(src + i);
            }
#if HAVE_TDM
            if (w == 0) __builtin_amdgcn_s_wait_tensorcnt(0);
#else
            // ---- W2 chunk: contiguous [n][32] block, flat coalesced copy ----
            {
                const unsigned short* srcc =
                    w2bf + (size_t)(e * NCHUNK + fc) * D_MODELc * 32;
#pragma unroll
                for (int it = 0; it < 16; ++it) {
                    int idx = tid * 8 + it * 2048;   // 8 bf16 per thread per step
                    uint4 v = *(const uint4*)(srcc + idx);
                    *(uint4*)(W2t + (idx >> 5) * W2LD + (idx & 31)) = v;
                }
            }
#endif
        } else {
            // ---- stage W1[:, ff0:ff0+32] transposed [n][k], convert f32->bf16 ----
            {
                const float* src = w1col + ff0;
#pragma unroll 4
                for (int k = skb; k < skb + 128; k += 8)
                    stage8_kstride(src + (size_t)k * D_FFc, D_FFc, w1dst + k);
            }
            // ---- stage W2[ff0:ff0+32, :] transposed [n][k], convert f32->bf16 ----
            {
                const float* base = w2e + (size_t)ff0 * D_MODELc;
#pragma unroll
                for (int c = 0; c < 4; ++c) {
                    int n = tid + 256 * c;
                    const float* src = base + n;
                    unsigned short* dst = W2t + n * W2LD;
#pragma unroll
                    for (int k = 0; k < 32; k += 8)
                        stage8_kstride(src + (size_t)k * D_MODELc, D_MODELc, dst + k);
                }
            }
        }
        __syncthreads();

        // ---- phase 1: partial H = X * W1chunk, half the K range per wave ----
        v8f p1 = vzero;
#pragma unroll
        for (int ks = 0; ks < 16; ++ks) {
            int k0 = (kh * 16 + ks) * 32;
            v16bf A = lds_frag(Xl,  arow, XLD, k0, laneHi8);
            v16bf B = lds_frag(W1t, brow, XLD, k0, laneHi8);
            p1 = wmma_bf16(A, B, p1);
        }
        {
            union { v8f v; float4 q[2]; } up; up.v = p1;
            float* pp = pb + (w * 32 + lane) * 8;
            *(float4*)(pp + 0) = up.q[0];
            *(float4*)(pp + 4) = up.q[1];
        }
        __syncthreads();

        // ---- combine K-halves, bias, silu, write H (bf16) ----
        if (w < 4) {
            const float* a0 = pb + ((w    ) * 32 + lane) * 8;
            const float* a1 = pb + ((w + 4) * 32 + lane) * 8;
            float bias = b1[(size_t)e * D_FFc + ff0 + (w & 1) * 16 + l15];
            int hrow0 = ((w >> 1) * 16) + laneHi8;
            int hcol  = (w & 1) * 16 + l15;
#pragma unroll
            for (int j = 0; j < 8; ++j) {
                float s2  = a0[j] + a1[j] + bias;
                float act = s2 / (1.f + __expf(-s2));          // silu
                Hl[(hrow0 + j) * HLD + hcol] = f32_to_bf16_bits(act);
            }
        }
        __syncthreads();

        // ---- phase 2: acc += H * W2t (pure LDS + WMMA) ----
        if (fc + 1 < NCHUNK) {
            if constexpr (PRECONV) {
                __builtin_prefetch(w1be + (size_t)(ff0 + 32) * D_MODELc + tid * 8, 0, 1);
                __builtin_prefetch(w2bf + (size_t)(e * NCHUNK + fc + 1) * D_MODELc * 32
                                   + tid * 8, 0, 1);
            } else {
                __builtin_prefetch(w1col + ff0 + 32 + (size_t)skb * D_FFc, 0, 1);
                __builtin_prefetch(w2e + (size_t)(ff0 + 32) * D_MODELc + tid * 4, 0, 1);
            }
        }
#pragma unroll
        for (int m = 0; m < 2; ++m) {
            v16bf Ah = lds_frag(Hl, m * 16 + l15, HLD, 0, laneHi8);
#pragma unroll
            for (int nt = 0; nt < 8; ++nt) {
                v16bf Bw = lds_frag(W2t, w * 128 + nt * 16 + l15, W2LD, 0, laneHi8);
                acc[m][nt] = wmma_bf16(Ah, Bw, acc[m][nt]);
            }
        }
        __syncthreads();   // protect W1t/W2t/pb/Hl reuse next chunk
    }

    // ---- epilogue: out[token] += cw * (acc + b2) ----
#pragma unroll
    for (int m = 0; m < 2; ++m) {
#pragma unroll
        for (int nt = 0; nt < 8; ++nt) {
            int col = w * 128 + nt * 16 + l15;
            float b2v = b2[(size_t)e * D_MODELc + col];
#pragma unroll
            for (int j = 0; j < 8; ++j) {
                int rl = m * 16 + j + laneHi8;
                float wv = swt[rl];
                if (wv != 0.f)
                    atomicAdd(out + (size_t)stk[rl] * D_MODELc + col,
                              wv * (acc[m][nt][j] + b2v));
            }
        }
    }
}

// ---------------- host entry ----------------

extern "C" void kernel_launch(void* const* d_in, const int* in_sizes, int n_in,
                              void* d_out, int out_size, void* d_ws, size_t ws_size,
                              hipStream_t stream) {
    (void)in_sizes; (void)n_in; (void)out_size;
    const float* x  = (const float*)d_in[0];
    const float* Wr = (const float*)d_in[1];
    const float* br = (const float*)d_in[2];
    const float* W1 = (const float*)d_in[3];
    const float* b1 = (const float*)d_in[4];
    const float* W2 = (const float*)d_in[5];
    const float* b2 = (const float*)d_in[6];
    float* out = (float*)d_out;

    // workspace: cw[T*E] | cnt[E] | probsum[E] | lists[E*T] | (w1bf | w2bf)
    char* ws = (char*)d_ws;
    float* cw      = (float*)(ws);
    int*   cnt     = (int*)  (ws + (size_t)TTOK * NEXP * 4);
    float* probsum = (float*)(ws + (size_t)TTOK * NEXP * 4 + 32);
    int*   lists   = (int*)  (ws + (size_t)TTOK * NEXP * 4 + 64);
    const size_t BOOK   = (size_t)TTOK * NEXP * 4 + 64 + (size_t)TTOK * NEXP * 4;
    const size_t WBYTES = (size_t)NEXP * D_MODELc * D_FFc * 2;   // 64 MB each
    unsigned short* w1bf = (unsigned short*)(ws + BOOK);
    unsigned short* w2bf = (unsigned short*)(ws + BOOK + WBYTES);
    bool preconv = ws_size >= BOOK + 2 * WBYTES;

    // 1) zero out + counters
    moe_init_kernel<<<dim3((TTOK * D_MODELc / 4) / 256), 256, 0, stream>>>(out, cnt, probsum);
    // 2) optional one-shot weight convert (f32 -> bf16, staging layouts)
    if (preconv) {
        moe_conv_w1_kernel<<<dim3(NEXP * D_FFc / 256), 256, 0, stream>>>(W1, w1bf);
        moe_conv_w2_kernel<<<dim3(NEXP * 4 * D_MODELc / 256), 256, 0, stream>>>(W2, w2bf);
    }
    // 3) router
    moe_router_kernel<<<dim3(TTOK / 256), 256, 0, stream>>>(x, Wr, br, cw, cnt, probsum, lists);
    // 4) aux loss scalar
    moe_aux_kernel<<<dim3(1), 32, 0, stream>>>(cnt, probsum, out);
    // 5) fused expert FFN (WMMA bf16)
    if (preconv) {
        moe_ffn_kernel<true><<<dim3(NEXP * (TTOK / 32)), 256, 0, stream>>>(
            x, W1, b1, W2, b2, cw, cnt, lists, out, w1bf, w2bf);
    } else {
        moe_ffn_kernel<false><<<dim3(NEXP * (TTOK / 32)), 256, 0, stream>>>(
            x, W1, b1, W2, b2, cw, cnt, lists, out, w1bf, w2bf);
    }
}
